// Gemma4PatchEmbed_3418793968084
// MI455X (gfx1250) — compile-verified, matrix operand order checked by reading, not run on hardware
//
#include <hip/hip_runtime.h>
#include <stdint.h>

typedef __bf16 bf16;
typedef bf16  v16bf __attribute__((ext_vector_type(16)));
typedef bf16  v8bf  __attribute__((ext_vector_type(8)));
typedef float v8f   __attribute__((ext_vector_type(8)));
typedef int   v4i   __attribute__((__vector_size__(16)));   // matches builtin param type

#define D_DIM 768
#define M_DIM 32768
#define V_DIM 10240
#define BM 128
#define BN 128
#define BK 32
#define LDT 40                 // padded LDS row stride in elements (80 bytes -> conflict-free)
#define NK (D_DIM / BK)        // 24 k-steps

#if defined(__has_builtin)
#if __has_builtin(__builtin_amdgcn_global_load_async_to_lds_b128) && __has_builtin(__builtin_amdgcn_s_wait_asynccnt)
#define USE_ASYNC_LDS 1
#endif
#endif

// Global (AS1) and LDS (AS3) typed pointer casts for the async-DMA builtin.
// AS3 pointers are 32-bit; the low 32 bits of a flat shared address are the LDS offset.
#define ASG(p) ((__attribute__((address_space(1))) v4i*)(uintptr_t)(p))
#define ASL(p) ((__attribute__((address_space(3))) v4i*)(uint32_t)(uintptr_t)(p))

// ---------------------------------------------------------------------------
// Prep: Wp[d][k] = bf16( 2 * W[d][ (k%3)*256 + k/3 ] ),  bias[d] = -sum_k W[d][k]
// Folds the (16,16,3)->(3,16,16) permutation AND the 2x-1 affine into the weights.
// ---------------------------------------------------------------------------
__global__ __launch_bounds__(256) void prep_w_kernel(const float* __restrict__ W,
                                                     bf16* __restrict__ Wp,
                                                     float* __restrict__ bias) {
    __shared__ float red[256];
    const int d = blockIdx.x;
    const int t = threadIdx.x;
    const float* row = W + (size_t)d * D_DIM;
    bf16* prow = Wp + (size_t)d * D_DIM;
    float s = 0.f;
    #pragma unroll
    for (int k = t; k < D_DIM; k += 256) {
        s += row[k];
        int src = (k % 3) * 256 + (k / 3);
        prow[k] = (bf16)(2.0f * row[src]);
    }
    red[t] = s;
    __syncthreads();
    for (int off = 128; off > 0; off >>= 1) {
        if (t < off) red[t] += red[t + off];
        __syncthreads();
    }
    if (t == 0) bias[d] = -red[0];
}

// ---------------------------------------------------------------------------
// ids / padding outputs: position_ids = coord[..., ::-1] (int64), pad = !valid
// ---------------------------------------------------------------------------
__global__ __launch_bounds__(256) void ids_kernel(const long long* __restrict__ pc,
                                                  const unsigned char* __restrict__ pv,
                                                  long long* __restrict__ ids,
                                                  unsigned char* __restrict__ pad) {
    int m = blockIdx.x * blockDim.x + threadIdx.x;
    if (m < M_DIM) {
        ids[2 * m + 0] = pc[2 * m + 1];
        ids[2 * m + 1] = pc[2 * m + 0];
        pad[m] = pv[m] ? 0 : 1;
    }
}

// ---------------------------------------------------------------------------
// GEMM: out[m,d] = sum_k x[m,k] * Wp[d,k] + bias[d] (+ pos gather, masked)
// 128x128x32 tiles, 8 waves, each wave 32x64 via v_wmma_f32_16x16x32_bf16.
// B tile via async global->LDS DMA (ASYNCcnt), A tile via regs (f32->bf16 cvt).
// ---------------------------------------------------------------------------
__global__ __launch_bounds__(256) void gemm_kernel(const float* __restrict__ x,
                                                   const bf16* __restrict__ Wp,
                                                   const float* __restrict__ bias,
                                                   const long long* __restrict__ pcoord,
                                                   const unsigned char* __restrict__ pvalid,
                                                   const float* __restrict__ ptab,
                                                   float* __restrict__ out) {
    __shared__ __align__(16) bf16 As[2][BM * LDT];
    __shared__ __align__(16) bf16 Bs[2][BN * LDT];
    __shared__ int sY[BM];
    __shared__ int sX[BM];
    __shared__ int sV[BM];

    const int tid = threadIdx.x;
    const int bn0 = blockIdx.x * BN;   // 6 tiles along D
    const int bm0 = blockIdx.y * BM;   // 256 tiles along M

    // Stage per-row coord/valid data for the epilogue.
    if (tid < BM) {
        int m = bm0 + tid;
        long long cy = pcoord[2 * m + 1];   // position_ids[...,0] = coord[...,1]
        long long cx = pcoord[2 * m + 0];
        sY[tid] = cy > 0 ? (int)cy : 0;
        sX[tid] = cx > 0 ? (int)cx : 0;
        sV[tid] = pvalid[m] ? 1 : 0;
    }

    // Tile-load mapping: thread t handles 16 contiguous k of one row.
    const int ldrow = tid >> 1;          // 0..127
    const int ldk   = (tid & 1) * 16;    // 0 or 16
    const float* aSrcBase = x  + (size_t)(bm0 + ldrow) * D_DIM + ldk;
    const bf16*  bSrcBase = Wp + (size_t)(bn0 + ldrow) * D_DIM + ldk;
    const int ldsOff = ldrow * LDT + ldk;

    float areg[16];

    auto issueA = [&](int k0) {
        const float4* p = (const float4*)(aSrcBase + k0);
        float4 f0 = p[0], f1 = p[1], f2 = p[2], f3 = p[3];
        areg[0]=f0.x;  areg[1]=f0.y;  areg[2]=f0.z;  areg[3]=f0.w;
        areg[4]=f1.x;  areg[5]=f1.y;  areg[6]=f1.z;  areg[7]=f1.w;
        areg[8]=f2.x;  areg[9]=f2.y;  areg[10]=f2.z; areg[11]=f2.w;
        areg[12]=f3.x; areg[13]=f3.y; areg[14]=f3.z; areg[15]=f3.w;
    };
    auto storeA = [&](int buf) {
        __align__(16) bf16 tmp[16];
        #pragma unroll
        for (int i = 0; i < 16; ++i) tmp[i] = (bf16)areg[i];
        *(v8bf*)&As[buf][ldsOff]     = *(const v8bf*)&tmp[0];
        *(v8bf*)&As[buf][ldsOff + 8] = *(const v8bf*)&tmp[8];
    };
    auto issueB = [&](int buf, int k0) {
#ifdef USE_ASYNC_LDS
        __builtin_amdgcn_global_load_async_to_lds_b128(
            ASG(bSrcBase + k0), ASL(&Bs[buf][ldsOff]), 0, 0);
        __builtin_amdgcn_global_load_async_to_lds_b128(
            ASG(bSrcBase + k0 + 8), ASL(&Bs[buf][ldsOff + 8]), 0, 0);
#else
        v4i q0 = *(const v4i*)(bSrcBase + k0);
        v4i q1 = *(const v4i*)(bSrcBase + k0 + 8);
        *(v4i*)&Bs[buf][ldsOff]     = q0;
        *(v4i*)&Bs[buf][ldsOff + 8] = q1;
#endif
    };

    // Wave tiling: 4 waves along M (32 rows each), 2 along N (64 cols each).
    const int lane = tid & 31;
    const int wid  = tid >> 5;
    const int wm   = (wid & 3) * 32;
    const int wn   = (wid >> 2) * 64;
    const int frow = lane & 15;          // row within 16x16 fragment
    const int fk   = (lane >> 4) * 16;   // K-half per lane group

    v8f acc[2][4];
    #pragma unroll
    for (int mi = 0; mi < 2; ++mi)
        #pragma unroll
        for (int ni = 0; ni < 4; ++ni)
            acc[mi][ni] = (v8f){0.f,0.f,0.f,0.f,0.f,0.f,0.f,0.f};

    // Prologue: fill buffer 0.
    issueA(0);
    issueB(0, 0);
    storeA(0);
#ifdef USE_ASYNC_LDS
    __builtin_amdgcn_s_wait_asynccnt(0);
#endif
    __syncthreads();

    for (int kt = 0; kt < NK; ++kt) {
        const int cur = kt & 1;
        if (kt + 1 < NK) {               // start next tile's loads (overlap with WMMA)
            issueA((kt + 1) * BK);
            issueB(cur ^ 1, (kt + 1) * BK);
        }

        v16bf af[2], bfr[4];
        #pragma unroll
        for (int mi = 0; mi < 2; ++mi) {
            int r = (wm + mi * 16 + frow) * LDT + fk;
            union { v16bf v; v8bf h[2]; } u;
            u.h[0] = *(const v8bf*)&As[cur][r];
            u.h[1] = *(const v8bf*)&As[cur][r + 8];
            af[mi] = u.v;
        }
        #pragma unroll
        for (int ni = 0; ni < 4; ++ni) {
            int r = (wn + ni * 16 + frow) * LDT + fk;
            union { v16bf v; v8bf h[2]; } u;
            u.h[0] = *(const v8bf*)&Bs[cur][r];
            u.h[1] = *(const v8bf*)&Bs[cur][r + 8];
            bfr[ni] = u.v;
        }
        #pragma unroll
        for (int mi = 0; mi < 2; ++mi)
            #pragma unroll
            for (int ni = 0; ni < 4; ++ni)
                acc[mi][ni] = __builtin_amdgcn_wmma_f32_16x16x32_bf16(
                    false, af[mi], false, bfr[ni], (short)0, acc[mi][ni], false, false);

        if (kt + 1 < NK) {
            storeA(cur ^ 1);
#ifdef USE_ASYNC_LDS
            __builtin_amdgcn_s_wait_asynccnt(0);
#endif
        }
        __syncthreads();
    }

    // Epilogue: bias + masked positional gather, fused.
    const float* t0 = ptab;
    const float* t1 = ptab + (size_t)V_DIM * D_DIM;
    #pragma unroll
    for (int mi = 0; mi < 2; ++mi) {
        #pragma unroll
        for (int ni = 0; ni < 4; ++ni) {
            const int d  = bn0 + wn + ni * 16 + frow;
            const float bd = bias[d];
            const int mbase = wm + mi * 16 + ((lane >> 4) << 3);
            v8f c = acc[mi][ni];
            #pragma unroll
            for (int r = 0; r < 8; ++r) {
                const int ml = mbase + r;
                float v = c[r] + bd;
                if (sV[ml])
                    v += t0[(size_t)sY[ml] * D_DIM + d] + t1[(size_t)sX[ml] * D_DIM + d];
                out[(size_t)(bm0 + ml) * D_DIM + d] = v;
            }
        }
    }
}

// ---------------------------------------------------------------------------
extern "C" void kernel_launch(void* const* d_in, const int* in_sizes, int n_in,
                              void* d_out, int out_size, void* d_ws, size_t ws_size,
                              hipStream_t stream) {
    (void)in_sizes; (void)n_in; (void)out_size; (void)ws_size;

    const float*         x      = (const float*)d_in[0];
    const long long*     pcoord = (const long long*)d_in[1];   // int64
    const unsigned char* pvalid = (const unsigned char*)d_in[2];
    const float*         projw  = (const float*)d_in[3];
    const float*         ptab   = (const float*)d_in[4];

    // Workspace: bias (768 f32) then permuted bf16 weights (768x768).
    float* bias = (float*)d_ws;
    bf16*  Wp   = (bf16*)((char*)d_ws + D_DIM * sizeof(float));

    // Output layout: [B,N,D] f32, then [B,N,2] int64, then [B,N] bool.
    float* outX = (float*)d_out;
    long long* ids = (long long*)(outX + (size_t)M_DIM * D_DIM);
    unsigned char* pad = (unsigned char*)(ids + (size_t)M_DIM * 2);

    prep_w_kernel<<<D_DIM, 256, 0, stream>>>(projw, Wp, bias);
    ids_kernel<<<M_DIM / 256, 256, 0, stream>>>(pcoord, pvalid, ids, pad);

    dim3 grid(D_DIM / BN, M_DIM / BM, 1);   // 6 x 256 tiles
    gemm_kernel<<<grid, 256, 0, stream>>>(x, Wp, bias, pcoord, pvalid, ptab, outX);
}
